// Qwen3MoeDecoderLayer_6073083756868
// MI455X (gfx1250) — matrix operand order repaired and checked by reading, not compiled, not run on hardware
//
#include <hip/hip_runtime.h>
#include <stdint.h>

typedef unsigned short u16;
typedef __attribute__((ext_vector_type(16))) __bf16 v16bf;
typedef __attribute__((ext_vector_type(8)))  float  v8f;
typedef __attribute__((ext_vector_type(4)))  uint32_t su4;
typedef __attribute__((ext_vector_type(8)))  uint32_t su8;

constexpr int B_ = 2, S_ = 1024, D_ = 2048, H_ = 16, KV_ = 4, DH_ = 128;
constexpr int E_ = 16, I_ = 1024;
constexpr int T_ = B_ * S_;          // 2048 tokens
constexpr int HD_ = H_ * DH_;        // 2048
constexpr int KD_ = KV_ * DH_;       // 512
constexpr int GROUPS_ = H_ / KV_;    // 4
constexpr float EPS_ = 1e-6f;
constexpr float THETA_ = 1000000.0f;

// ---------------- helpers ----------------

__device__ inline u16 f2bf(float x) {
  uint32_t u = __float_as_uint(x);
  u += 0x7fffu + ((u >> 16) & 1u);   // round-to-nearest-even
  return (u16)(u >> 16);
}

__device__ inline v16bf load_frag2(const u16* p0, const u16* p1) {
  union { v16bf v; uint4 q[2]; } f;
  f.q[0] = *reinterpret_cast<const uint4*>(p0);
  f.q[1] = *reinterpret_cast<const uint4*>(p1);
  return f.v;
}

// A-matrix fragment (16x32 bf16, row-major A with leading dim lda).
// ISA layout: lanes 0-15 hold M=lane, K=0..7 & 16..23; lanes 16-31 hold K=8..15 & 24..31.
__device__ inline v16bf load_a_frag(const u16* A, int lda, int m0, int k0) {
  const int lane = threadIdx.x & 31;
  const u16* p = A + (size_t)(m0 + (lane & 15)) * lda + k0 + ((lane >> 4) << 3);
  return load_frag2(p, p + 16);
}

// B-matrix fragment (32x16 bf16) from B stored transposed as [N][K] with leading dim ldb.
__device__ inline v16bf load_b_frag(const u16* Bt, int ldb, int n0, int k0) {
  const int lane = threadIdx.x & 31;
  const u16* p = Bt + (size_t)(n0 + (lane & 15)) * ldb + k0 + ((lane >> 4) << 3);
  return load_frag2(p, p + 16);
}

__device__ inline v8f wmma_bf16(v16bf a, v16bf b, v8f c) {
  return __builtin_amdgcn_wmma_f32_16x16x32_bf16(false, a, false, b, (short)0, c,
                                                 false, false);
}

// ---------------- Tensor Data Mover: 2D bf16 tile Global -> LDS ----------------
// Builds a D# (group0 + group1) per CDNA5 ISA and issues
// TENSOR_LOAD_TO_LDS. Tracked by TENSORcnt. One wave issues per workgroup.

__device__ inline void tdm_load_2d_bf16(uint32_t lds_byte_addr, const u16* gsrc,
                                        uint32_t kdim, uint32_t mdim,
                                        uint32_t tile_k, uint32_t tile_m,
                                        uint32_t stride_elems) {
  const uint64_t ga = (uint64_t)(uintptr_t)gsrc;
  su4 g0;
  g0[0] = 0x1u;                                    // count=1, user mode, no gather
  g0[1] = lds_byte_addr;                           // lds_addr
  g0[2] = (uint32_t)ga;                            // global_addr[31:0]
  g0[3] = (uint32_t)((ga >> 32) & 0x01FFFFFFu)     // global_addr[56:32]
          | (2u << 30);                            // type=2 ("image")
  su8 g1;
  g1[0] = 1u << 16;                                // data_size=1 (2 bytes/elem)
  g1[1] = (kdim & 0xFFFFu) << 16;                  // tensor_dim0[15:0]
  g1[2] = (kdim >> 16) | ((mdim & 0xFFFFu) << 16); // tensor_dim0[31:16] | tensor_dim1[15:0]
  g1[3] = (mdim >> 16) | (tile_k << 16);           // tensor_dim1[31:16] | tile_dim0
  g1[4] = tile_m;                                  // tile_dim1 (tile_dim2=0)
  g1[5] = stride_elems;                            // tensor_dim0_stride[31:0]
  g1[6] = 0;                                       // dim0_stride[47:32] | dim1_stride[15:0]
  g1[7] = 0;                                       // tensor_dim1_stride[47:16]
  asm volatile("tensor_load_to_lds %0, %1" :: "s"(g0), "s"(g1) : "memory");
}

// ---------------- weight convert + transpose (fp32 [R][C] -> bf16 [C][R]) ----------------

__global__ void __launch_bounds__(256)
cvt_t_kernel(const float* __restrict__ in, u16* __restrict__ out, int R, int C) {
  const size_t nel = (size_t)R * C;
  size_t idx = (size_t)blockIdx.x * 256 + threadIdx.x;
  if (idx >= nel) return;
  const float* ib = in + (size_t)blockIdx.y * nel;
  u16* ob = out + (size_t)blockIdx.y * nel;
  size_t r = idx / C, c = idx % C;
  ob[c * (size_t)R + r] = f2bf(ib[idx]);
}

// ---------------- RMSNorm (fp32 in, bf16 out) ----------------

__global__ void __launch_bounds__(256)
rmsnorm_kernel(const float* __restrict__ x, const float* __restrict__ w,
               u16* __restrict__ out, int Dm) {
  const int row = blockIdx.x;
  const float* xr = x + (size_t)row * Dm;
  float ss = 0.f;
  for (int i = threadIdx.x; i < Dm; i += 256) { float v = xr[i]; ss += v * v; }
#pragma unroll
  for (int off = 16; off; off >>= 1) ss += __shfl_xor(ss, off, 32);
  __shared__ float red[8];
  if ((threadIdx.x & 31) == 0) red[threadIdx.x >> 5] = ss;
  __syncthreads();
  float tot = 0.f;
#pragma unroll
  for (int i = 0; i < 8; ++i) tot += red[i];
  const float inv = rsqrtf(tot / (float)Dm + EPS_);
  u16* orow = out + (size_t)row * Dm;
  for (int i = threadIdx.x; i < Dm; i += 256) orow[i] = f2bf(xr[i] * inv * w[i]);
}

// ---------------- bf16 WMMA GEMM with TDM-staged A tiles ----------------
// C[M,N] = A[M,K] * Bt[N,K]^T (+resid). Block = 4 waves, 16x64 output tile.
// The shared 16xKC A-chunk is DMA'd into LDS by the Tensor Data Mover (wave 0),
// double-buffered against WMMA compute; B (weights) streams from global.
// Requires K % 256 == 0 (all call sites satisfy this).

constexpr int KC_ = 256;

__global__ void __launch_bounds__(128)
gemm_bf16_kernel(const u16* __restrict__ A, int lda,
                 const u16* __restrict__ Bt, int ldb,
                 float* __restrict__ C, int ldc,
                 int M, int N, int K,
                 const float* __restrict__ resid) {
  __shared__ __align__(16) u16 atile[2][16 * KC_];   // 2 x 8KB double buffer
  const int wave = threadIdx.x >> 5;
  const int m0 = blockIdx.x * 16;
  const int n0 = blockIdx.y * 64 + wave * 16;
  const bool active_n = (n0 < N);
  const int n0c = active_n ? n0 : 0;
  const int lane = threadIdx.x & 31;
  const int half = lane >> 4, col = lane & 15;

  // prologue: TDM chunk 0
  if (threadIdx.x < 32) {
    tdm_load_2d_bf16((uint32_t)(uintptr_t)&atile[0][0],
                     A + (size_t)m0 * lda, (uint32_t)K, (uint32_t)M,
                     KC_, 16, (uint32_t)lda);
    __builtin_amdgcn_s_wait_tensorcnt(0);
  }
  __syncthreads();

  v8f acc = {};
  const int nchunk = K / KC_;
  for (int ch = 0; ch < nchunk; ++ch) {
    const int cur = ch & 1;
    if (ch + 1 < nchunk && threadIdx.x < 32)
      tdm_load_2d_bf16((uint32_t)(uintptr_t)&atile[cur ^ 1][0],
                       A + (size_t)m0 * lda + (ch + 1) * KC_,
                       (uint32_t)K, (uint32_t)M, KC_, 16, (uint32_t)lda);
    const u16* As = &atile[cur][0];
#pragma unroll
    for (int kk = 0; kk < KC_; kk += 32) {
      v16bf a = load_a_frag(As, KC_, 0, kk);
      v16bf b = load_b_frag(Bt, ldb, n0c, ch * KC_ + kk);
      if (ch * KC_ + kk + 256 < K)
        __builtin_prefetch(Bt + (size_t)(n0c + col) * ldb + ch * KC_ + kk + 256, 0, 1);
      acc = wmma_bf16(a, b, acc);
    }
    if (ch + 1 < nchunk) {
      if (threadIdx.x < 32) __builtin_amdgcn_s_wait_tensorcnt(0);
      __syncthreads();   // TDM chunk ch+1 visible; buffer `cur` free for reuse
    }
  }

  if (active_n) {
#pragma unroll
    for (int r = 0; r < 8; ++r) {
      const int m = m0 + r + 8 * half;
      const size_t idx = (size_t)m * ldc + n0 + col;
      float v = acc[r];
      if (resid) v += resid[idx];
      C[idx] = v;
    }
  }
}

// ---------------- per-head QK RMSNorm + RoPE (fp32 in, bf16 out) ----------------

__global__ void __launch_bounds__(128)
qknorm_rope_kernel(const float* __restrict__ in, const float* __restrict__ nw,
                   const int* __restrict__ pos, u16* __restrict__ out, int NH) {
  const int token = blockIdx.x;  // b*S + s
  const int h = blockIdx.y;
  const int d = threadIdx.x;     // 0..127
  const size_t base = ((size_t)token * NH + h) * DH_;
  const float v = in[base + d];
  float ss = v * v;
#pragma unroll
  for (int off = 16; off; off >>= 1) ss += __shfl_xor(ss, off, 32);
  __shared__ float red[4];
  if ((d & 31) == 0) red[d >> 5] = ss;
  __syncthreads();
  const float tot = red[0] + red[1] + red[2] + red[3];
  const float xv = v * rsqrtf(tot / (float)DH_ + EPS_) * nw[d];
  __shared__ float sh[DH_];
  sh[d] = xv;
  __syncthreads();
  const float p = (float)pos[token];
  const float freq = __powf(THETA_, -(float)(2 * (d & 63)) / (float)DH_);
  const float ang = p * freq;
  const float rot = (d < 64) ? -sh[d + 64] : sh[d - 64];
  out[base + d] = f2bf(xv * __cosf(ang) + rot * __sinf(ang));
}

// ---------------- V: fp32 [B,S,KV,DH] -> bf16 transposed [B,KV,DH,S] ----------------

__global__ void __launch_bounds__(128)
vcvt_kernel(const float* __restrict__ vin, u16* __restrict__ vt) {
  const int token = blockIdx.x, kv = blockIdx.y, d = threadIdx.x;
  const int b = token / S_, s = token % S_;
  const float x = vin[((size_t)token * KV_ + kv) * DH_ + d];
  vt[(((size_t)b * KV_ + kv) * DH_ + d) * S_ + s] = f2bf(x);
}

// ---------------- flash attention: 1 wave per 16-query tile, causal GQA ----------------

__global__ void __launch_bounds__(32)
flash_attn_kernel(const u16* __restrict__ qb, const u16* __restrict__ kb,
                  const u16* __restrict__ vtb, u16* __restrict__ ob) {
  const int qt = blockIdx.x, h = blockIdx.y, b = blockIdx.z;
  const int kvh = h / GROUPS_;
  const int lane = threadIdx.x & 31;
  const int half = lane >> 4, col = lane & 15;
  const int m0 = qt * 16;
  const float scale = 0.08838834764831845f;  // 1/sqrt(128)

  const u16* Aq = qb + ((size_t)b * S_ * H_ + h) * DH_;          // lda = HD_
  const u16* Bk = kb + ((size_t)b * S_ * KV_ + kvh) * DH_;       // ldb = KD_
  const u16* Bv = vtb + (size_t)(b * KV_ + kvh) * DH_ * S_;      // ldb = S_

  v16bf qa[4];
#pragma unroll
  for (int c = 0; c < 4; ++c) qa[c] = load_a_frag(Aq, HD_, m0, c * 32);

  v8f o[8];
#pragma unroll
  for (int c = 0; c < 8; ++c) o[c] = (v8f){};
  float mrow[8], lrow[8];
#pragma unroll
  for (int r = 0; r < 8; ++r) { mrow[r] = -3.0e38f; lrow[r] = 0.f; }

  __shared__ __align__(16) u16 pt[16 * 32];

  for (int j = 0; j < m0 + 16; j += 32) {
    v8f s0 = {}, s1 = {};
#pragma unroll
    for (int c = 0; c < 4; ++c) {
      s0 = wmma_bf16(qa[c], load_b_frag(Bk, KD_, j, c * 32), s0);
      s1 = wmma_bf16(qa[c], load_b_frag(Bk, KD_, j + 16, c * 32), s1);
    }
    float alpha[8];
#pragma unroll
    for (int r = 0; r < 8; ++r) {
      const int mr = r + 8 * half;
      const int m = m0 + mr;
      float x0 = s0[r] * scale, x1v = s1[r] * scale;
      if (j + col > m)      x0 = -3.0e38f;
      if (j + 16 + col > m) x1v = -3.0e38f;
      float mx = fmaxf(x0, x1v);
#pragma unroll
      for (int off = 1; off < 16; off <<= 1) mx = fmaxf(mx, __shfl_xor(mx, off, 32));
      const float mnew = fmaxf(mrow[r], mx);
      const float p0 = __expf(x0 - mnew);
      const float p1 = __expf(x1v - mnew);
      float rs = p0 + p1;
#pragma unroll
      for (int off = 1; off < 16; off <<= 1) rs += __shfl_xor(rs, off, 32);
      alpha[r] = __expf(mrow[r] - mnew);
      lrow[r] = lrow[r] * alpha[r] + rs;
      mrow[r] = mnew;
      pt[mr * 32 + col] = f2bf(p0);
      pt[mr * 32 + col + 16] = f2bf(p1);
    }
#pragma unroll
    for (int c = 0; c < 8; ++c) {
      v8f t = o[c];
#pragma unroll
      for (int r = 0; r < 8; ++r) t[r] *= alpha[r];
      o[c] = t;
    }
    __syncthreads();
    v16bf pa = load_a_frag(pt, 32, 0, 0);
#pragma unroll
    for (int c = 0; c < 8; ++c)
      o[c] = wmma_bf16(pa, load_b_frag(Bv, S_, c * 16, j), o[c]);
    __syncthreads();
  }

  u16* Ob = ob + ((size_t)b * S_ * H_ + h) * DH_;
#pragma unroll
  for (int c = 0; c < 8; ++c)
#pragma unroll
    for (int r = 0; r < 8; ++r) {
      const int m = m0 + r + 8 * half;
      Ob[(size_t)m * HD_ + c * 16 + col] = f2bf(o[c][r] / lrow[r]);
    }
}

// ---------------- router softmax + top-2 -> dense per-token expert weights ----------------

__global__ void __launch_bounds__(256)
topk_kernel(const float* __restrict__ logits, float* __restrict__ wd) {
  const int t = blockIdx.x * 256 + threadIdx.x;
  if (t >= T_) return;
  float l[E_];
  float mx = -3.0e38f;
#pragma unroll
  for (int e = 0; e < E_; ++e) { l[e] = logits[t * E_ + e]; mx = fmaxf(mx, l[e]); }
#pragma unroll
  for (int e = 0; e < E_; ++e) l[e] = __expf(l[e] - mx);
  int i1 = 0; float v1 = -1.f;
#pragma unroll
  for (int e = 0; e < E_; ++e) if (l[e] > v1) { v1 = l[e]; i1 = e; }
  int i2 = 0; float v2 = -1.f;
#pragma unroll
  for (int e = 0; e < E_; ++e) if (e != i1 && l[e] > v2) { v2 = l[e]; i2 = e; }
  const float inv = 1.f / (v1 + v2);   // softmax denom cancels in top-k renorm
#pragma unroll
  for (int e = 0; e < E_; ++e) wd[t * E_ + e] = 0.f;
  wd[t * E_ + i1] = v1 * inv;
  wd[t * E_ + i2] = v2 * inv;
}

// ---------------- MoE gate/up: act[e,t,i] = silu(g)*u*w, tile-skip on zero weights ----------------

__global__ void __launch_bounds__(128)
moe_gateup_kernel(const u16* __restrict__ h2, const u16* __restrict__ gateT,
                  const u16* __restrict__ upT, const float* __restrict__ wd,
                  u16* __restrict__ act) {
  const int t0 = blockIdx.x * 16;
  const int e = blockIdx.y;
  __shared__ float wsm[16];
  if (threadIdx.x < 16) wsm[threadIdx.x] = wd[(size_t)(t0 + threadIdx.x) * E_ + e];
  __syncthreads();
  bool any = false;
#pragma unroll
  for (int i = 0; i < 16; ++i) any = any || (wsm[i] != 0.f);
  if (!any) return;  // top-2 sparsity: skip unrouted (expert, token-tile)
  const int wave = threadIdx.x >> 5, lane = threadIdx.x & 31;
  const int half = lane >> 4, col = lane & 15;
  const u16* gB = gateT + (size_t)e * I_ * D_;
  const u16* uB = upT + (size_t)e * I_ * D_;
  for (int nb = 0; nb < I_; nb += 64) {
    const int n0 = nb + wave * 16;
    v8f g = {}, u = {};
    for (int k0 = 0; k0 < D_; k0 += 32) {
      v16bf a = load_a_frag(h2, D_, t0, k0);
      g = wmma_bf16(a, load_b_frag(gB, D_, n0, k0), g);
      u = wmma_bf16(a, load_b_frag(uB, D_, n0, k0), u);
    }
#pragma unroll
    for (int r = 0; r < 8; ++r) {
      const int m = r + 8 * half;
      const float gv = g[r];
      const float a = (gv / (1.f + __expf(-gv))) * u[r] * wsm[m];
      act[((size_t)e * T_ + (t0 + m)) * I_ + n0 + col] = f2bf(a);
    }
  }
}

// ---------------- MoE down-proj + both residual adds -> final output ----------------

__global__ void __launch_bounds__(128)
moe_down_kernel(const u16* __restrict__ act, const u16* __restrict__ downT,
                const float* __restrict__ wd, const float* __restrict__ x1,
                float* __restrict__ out) {
  const int t0 = blockIdx.x * 16;
  const int wave = threadIdx.x >> 5, lane = threadIdx.x & 31;
  const int half = lane >> 4, col = lane & 15;
  const int n0 = blockIdx.y * 64 + wave * 16;
  __shared__ float wall[256];
  for (int i = threadIdx.x; i < 256; i += 128)
    wall[i] = wd[(size_t)(t0 + (i >> 4)) * E_ + (i & 15)];
  __syncthreads();
  v8f acc = {};
  for (int e = 0; e < E_; ++e) {
    bool any = false;
#pragma unroll
    for (int m = 0; m < 16; ++m) any = any || (wall[m * 16 + e] != 0.f);
    if (!any) continue;
    const u16* aB = act + ((size_t)e * T_ + t0) * I_;
    const u16* dB = downT + (size_t)e * D_ * I_;
    for (int k0 = 0; k0 < I_; k0 += 32)
      acc = wmma_bf16(load_a_frag(aB, I_, 0, k0), load_b_frag(dB, I_, n0, k0), acc);
  }
#pragma unroll
  for (int r = 0; r < 8; ++r) {
    const int m = t0 + r + 8 * half;
    const size_t idx = (size_t)m * D_ + n0 + col;
    out[idx] = x1[idx] + acc[r];
  }
}

// ---------------- host orchestration ----------------

extern "C" void kernel_launch(void* const* d_in, const int* in_sizes, int n_in,
                              void* d_out, int out_size, void* d_ws, size_t ws_size,
                              hipStream_t stream) {
  (void)in_sizes; (void)n_in; (void)out_size; (void)ws_size;
  const float* hidden = (const float*)d_in[0];
  const int* pos = (const int*)d_in[1];
  const float* ln1_w = (const float*)d_in[2];
  const float* q_w = (const float*)d_in[3];
  const float* k_w = (const float*)d_in[4];
  const float* v_w = (const float*)d_in[5];
  const float* o_w = (const float*)d_in[6];
  const float* qn_w = (const float*)d_in[7];
  const float* kn_w = (const float*)d_in[8];
  const float* ln2_w = (const float*)d_in[9];
  const float* rt_w = (const float*)d_in[10];
  const float* gate_w = (const float*)d_in[11];
  const float* up_w = (const float*)d_in[12];
  const float* down_w = (const float*)d_in[13];
  float* out = (float*)d_out;

  char* ws = (char*)d_ws;
  size_t off = 0;
  auto alloc = [&](size_t bytes) -> void* {
    void* p = ws + off;
    off += (bytes + 255) & ~(size_t)255;
    return p;
  };
  u16* qT = (u16*)alloc((size_t)HD_ * D_ * 2);
  u16* kTw = (u16*)alloc((size_t)KD_ * D_ * 2);
  u16* vTw = (u16*)alloc((size_t)KD_ * D_ * 2);
  u16* oT = (u16*)alloc((size_t)D_ * HD_ * 2);
  u16* rT = (u16*)alloc((size_t)E_ * D_ * 2);
  u16* gateT = (u16*)alloc((size_t)E_ * I_ * D_ * 2);
  u16* upT = (u16*)alloc((size_t)E_ * I_ * D_ * 2);
  u16* downT = (u16*)alloc((size_t)E_ * D_ * I_ * 2);
  u16* h1 = (u16*)alloc((size_t)T_ * D_ * 2);
  float* qf = (float*)alloc((size_t)T_ * HD_ * 4);
  float* kf = (float*)alloc((size_t)T_ * KD_ * 4);
  float* vf = (float*)alloc((size_t)T_ * KD_ * 4);
  u16* qbf = (u16*)alloc((size_t)T_ * HD_ * 2);
  u16* kbf = (u16*)alloc((size_t)T_ * KD_ * 2);
  u16* vtbf = (u16*)alloc((size_t)T_ * KD_ * 2);
  u16* attnbf = (u16*)alloc((size_t)T_ * HD_ * 2);
  float* x1 = (float*)alloc((size_t)T_ * D_ * 4);
  u16* h2 = (u16*)alloc((size_t)T_ * D_ * 2);
  float* logits = (float*)alloc((size_t)T_ * E_ * 4);
  float* wdense = (float*)alloc((size_t)T_ * E_ * 4);
  u16* actbuf = (u16*)alloc((size_t)E_ * T_ * I_ * 2);

  auto cdiv = [](size_t a, size_t b) { return (unsigned)((a + b - 1) / b); };

  // one-time weight bf16 conversion + transpose to [N][K]
  cvt_t_kernel<<<dim3(cdiv((size_t)D_ * HD_, 256), 1), 256, 0, stream>>>(q_w, qT, D_, HD_);
  cvt_t_kernel<<<dim3(cdiv((size_t)D_ * KD_, 256), 1), 256, 0, stream>>>(k_w, kTw, D_, KD_);
  cvt_t_kernel<<<dim3(cdiv((size_t)D_ * KD_, 256), 1), 256, 0, stream>>>(v_w, vTw, D_, KD_);
  cvt_t_kernel<<<dim3(cdiv((size_t)HD_ * D_, 256), 1), 256, 0, stream>>>(o_w, oT, HD_, D_);
  cvt_t_kernel<<<dim3(cdiv((size_t)D_ * E_, 256), 1), 256, 0, stream>>>(rt_w, rT, D_, E_);
  cvt_t_kernel<<<dim3(cdiv((size_t)D_ * I_, 256), E_), 256, 0, stream>>>(gate_w, gateT, D_, I_);
  cvt_t_kernel<<<dim3(cdiv((size_t)D_ * I_, 256), E_), 256, 0, stream>>>(up_w, upT, D_, I_);
  cvt_t_kernel<<<dim3(cdiv((size_t)I_ * D_, 256), E_), 256, 0, stream>>>(down_w, downT, I_, D_);

  // attention block
  rmsnorm_kernel<<<T_, 256, 0, stream>>>(hidden, ln1_w, h1, D_);
  gemm_bf16_kernel<<<dim3(T_ / 16, HD_ / 64), 128, 0, stream>>>(h1, D_, qT, D_, qf, HD_, T_, HD_, D_, nullptr);
  gemm_bf16_kernel<<<dim3(T_ / 16, KD_ / 64), 128, 0, stream>>>(h1, D_, kTw, D_, kf, KD_, T_, KD_, D_, nullptr);
  gemm_bf16_kernel<<<dim3(T_ / 16, KD_ / 64), 128, 0, stream>>>(h1, D_, vTw, D_, vf, KD_, T_, KD_, D_, nullptr);
  qknorm_rope_kernel<<<dim3(T_, H_), 128, 0, stream>>>(qf, qn_w, pos, qbf, H_);
  qknorm_rope_kernel<<<dim3(T_, KV_), 128, 0, stream>>>(kf, kn_w, pos, kbf, KV_);
  vcvt_kernel<<<dim3(T_, KV_), 128, 0, stream>>>(vf, vtbf);
  flash_attn_kernel<<<dim3(S_ / 16, H_, B_), 32, 0, stream>>>(qbf, kbf, vtbf, attnbf);
  gemm_bf16_kernel<<<dim3(T_ / 16, D_ / 64), 128, 0, stream>>>(attnbf, HD_, oT, HD_, x1, D_, T_, D_, HD_, hidden);

  // MoE block
  rmsnorm_kernel<<<T_, 256, 0, stream>>>(x1, ln2_w, h2, D_);
  gemm_bf16_kernel<<<dim3(T_ / 16, 1), 128, 0, stream>>>(h2, D_, rT, D_, logits, E_, T_, E_, D_, nullptr);
  topk_kernel<<<cdiv(T_, 256), 256, 0, stream>>>(logits, wdense);
  moe_gateup_kernel<<<dim3(T_ / 16, E_), 128, 0, stream>>>(h2, gateT, upT, wdense, actbuf);
  moe_down_kernel<<<dim3(T_ / 16, D_ / 64), 128, 0, stream>>>(actbuf, downT, wdense, x1, out);
}
